// _NaiveMPNN_77979426226444
// MI455X (gfx1250) — matrix operand order repaired and checked by reading, not compiled, not run on hardware
//
#include <hip/hip_runtime.h>
#include <hip/hip_bf16.h>
#include <stdint.h>

// ---------------- problem constants ----------------
constexpr int NV   = 4096;
constexpr int NE   = 8192;
constexpr int E2   = 2 * NE;     // 16384 directed edges
constexpr int HV   = 256;
constexpr int HE   = 128;
constexpr int KATT = HV + HE + HV;   // 640
constexpr int G3   = 3 * HV;         // 768
constexpr int OUTD = 256;

// ---------------- vector types for WMMA ----------------
typedef __bf16 v16bf __attribute__((ext_vector_type(16)));
typedef __bf16 v8bf  __attribute__((ext_vector_type(8)));
typedef float  v8f   __attribute__((ext_vector_type(8)));

// ---------------- workspace layout ----------------
constexpr size_t ALN = 256;
constexpr size_t al(size_t x) { return (x + ALN - 1) & ~(ALN - 1); }

constexpr size_t O_SRC   = 0;
constexpr size_t O_DST   = O_SRC  + al((size_t)E2 * 4);
constexpr size_t O_EW    = O_DST  + al((size_t)E2 * 4);
constexpr size_t O_DR    = O_EW   + al((size_t)E2 * 4);
constexpr size_t O_DC    = O_DR   + (size_t)NV * 4;          // contiguous with DR
constexpr size_t O_AW1P  = O_DC   + al((size_t)NV * 4);
constexpr size_t O_GWP   = O_AW1P + al((size_t)KATT * HV * 2);
constexpr size_t O_WIHP  = O_GWP  + al((size_t)4 * HV * HV * 2);
constexpr size_t O_WHHP  = O_WIHP + al((size_t)HV * G3 * 2);
constexpr size_t O_RW1P  = O_WHHP + al((size_t)HV * G3 * 2);
constexpr size_t O_RW2P  = O_RW1P + al((size_t)(2 * HV) * HV * 2);
constexpr size_t O_ADJ   = O_RW2P + al((size_t)HV * OUTD * 2);
constexpr size_t O_ADJBF = O_ADJ  + al((size_t)NV * NV * 4);
constexpr size_t O_HID   = O_ADJBF+ al((size_t)NV * NV * 2);
constexpr size_t O_HIDA  = O_HID  + al((size_t)NV * HV * 4);  // bf16 row-major
constexpr size_t O_HIDP  = O_HIDA + al((size_t)NV * HV * 2);  // bf16 B-packed
constexpr size_t O_T     = O_HIDP + al((size_t)NV * HV * 2);
constexpr size_t O_TBF   = O_T    + al((size_t)NV * HV * 4);
constexpr size_t O_MSG   = O_TBF  + al((size_t)NV * HV * 2);
constexpr size_t O_MSGBF = O_MSG  + al((size_t)NV * HV * 4);
constexpr size_t O_UNION = O_MSGBF+ al((size_t)NV * HV * 2);
// attention phase
constexpr size_t O_UEV   = O_UNION;
constexpr size_t O_HATT  = O_UEV  + al((size_t)E2 * KATT * 2);
// GRU phase (reuses union region)
constexpr size_t O_GI    = O_UNION;
constexpr size_t O_GH    = O_GI   + al((size_t)NV * G3 * 4);
// remap phase (reuses union region)
constexpr size_t O_CAT   = O_UNION;
constexpr size_t O_R1    = O_CAT  + al((size_t)NV * 2 * HV * 2);
constexpr size_t O_R1BF  = O_R1   + al((size_t)NV * HV * 4);

// ---------------- helpers ----------------
__device__ __forceinline__ unsigned short f2bf(float f) {
    unsigned u = __float_as_uint(f);
    unsigned r = (u + 0x7FFFu + ((u >> 16) & 1u)) >> 16;
    return (unsigned short)r;
}
__device__ __forceinline__ float sigm(float x) { return 1.f / (1.f + __expf(-x)); }

// ---------------- index extraction from one-hot incidence ----------------
__global__ void k_extract(const float* __restrict__ vew1, const float* __restrict__ vew2,
                          int* __restrict__ src2, int* __restrict__ dst2) {
    long long idx = (long long)blockIdx.x * 256 + threadIdx.x;   // = r*NE + e
    int e = (int)(idx & (NE - 1));
    int r = (int)(idx >> 13);                                    // NE = 2^13
    if (vew1[idx] != 0.0f) { src2[e] = r; dst2[NE + e] = r; }    // vew_u=[vew1|vew2]
    if (vew2[idx] != 0.0f) { dst2[e] = r; src2[NE + e] = r; }    // vew_v=[vew2|vew1]
}

// ---------------- uev = [hv[src] | he | hv[dst]] in bf16 ----------------
__global__ void k_uev(const float* __restrict__ hv, const float* __restrict__ he,
                      const int* __restrict__ src2, const int* __restrict__ dst2,
                      unsigned short* __restrict__ uev) {
    long long t = (long long)blockIdx.x * 256 + threadIdx.x;     // E2*KATT exact
    int c = (int)(t % KATT);
    int e = (int)(t / KATT);
    float v;
    if (c < HV)            v = hv[(size_t)src2[e] * HV + c];
    else if (c < HV + HE)  v = he[(size_t)(e & (NE - 1)) * HE + (c - HV)];
    else                   v = hv[(size_t)dst2[e] * HV + (c - HV - HE)];
    uev[t] = f2bf(v);
}

// ---------------- pack B into per-lane WMMA fragment layout ----------------
// dword d = ((k/32)*N + n)*16 + kh*8 + j  holds B(k', n), B(k'+1, n)
// with k' = (k/32)*32 + kh*16 + 2j.  One v16bf (32B) load = a full B fragment half.
__global__ void k_pack_b(const float* __restrict__ W, unsigned* __restrict__ Bp,
                         int K, int N, int trans) {
    int d = blockIdx.x * 256 + threadIdx.x;
    int total = (K >> 5) * N * 16;
    if (d >= total) return;
    int j = d & 7, khh = (d >> 3) & 1;
    int rest = d >> 4;
    int n = rest % N, t = rest / N;
    int k = (t << 5) + khh * 16 + (j << 1);
    float v0 = trans ? W[(size_t)n * K + k]     : W[(size_t)k * N + n];
    float v1 = trans ? W[(size_t)n * K + k + 1] : W[(size_t)(k + 1) * N + n];
    Bp[d] = (unsigned)f2bf(v0) | ((unsigned)f2bf(v1) << 16);
}

// ---------------- generic bf16 WMMA GEMM ----------------
// C[M,N] = act( A_bf16[M,K] @ Bpacked[K,N] + bias ), optional bf16 copy of C.
// block = 256 thr = 8 waves arranged 2x4; wave tile 32x32 (4 accumulators).
__device__ __forceinline__ v16bf load_a_frag(const unsigned short* A, long long row,
                                             int K, int k, int kh) {
    const v8bf lo = *reinterpret_cast<const v8bf*>(A + row * K + k + kh * 16);
    const v8bf hi = *reinterpret_cast<const v8bf*>(A + row * K + k + 8 + kh * 16);
    return __builtin_shufflevector(lo, hi, 0,1,2,3,4,5,6,7,8,9,10,11,12,13,14,15);
}

__global__ __launch_bounds__(256)
void k_gemm_bf16(const unsigned short* __restrict__ A,
                 const unsigned short* __restrict__ Bp,
                 const float* __restrict__ bias,
                 float* __restrict__ C,
                 unsigned short* __restrict__ Cbf,
                 int M, int N, int K, int act) {
    const int lane = threadIdx.x & 31;
    const int w    = threadIdx.x >> 5;
    const int l15  = lane & 15;
    const int kh   = lane >> 4;
    const int m0 = blockIdx.y * 64  + (w >> 2) * 32;
    const int n0 = blockIdx.x * 128 + (w & 3) * 32;

    v8f acc[2][2] = {};
    const int nsteps = K >> 5;
    for (int t = 0; t < nsteps; ++t) {
        const int k = t << 5;
        v16bf a0 = load_a_frag(A, (long long)(m0 + l15),      K, k, kh);
        v16bf a1 = load_a_frag(A, (long long)(m0 + 16 + l15), K, k, kh);
        const unsigned short* bb = Bp + (size_t)t * N * 32 + (size_t)kh * 16;
        v16bf b0 = *reinterpret_cast<const v16bf*>(bb + (size_t)(n0 + l15) * 32);
        v16bf b1 = *reinterpret_cast<const v16bf*>(bb + (size_t)(n0 + 16 + l15) * 32);
        acc[0][0] = __builtin_amdgcn_wmma_f32_16x16x32_bf16(false, a0, false, b0, (short)0, acc[0][0], false, false);
        acc[0][1] = __builtin_amdgcn_wmma_f32_16x16x32_bf16(false, a0, false, b1, (short)0, acc[0][1], false, false);
        acc[1][0] = __builtin_amdgcn_wmma_f32_16x16x32_bf16(false, a1, false, b0, (short)0, acc[1][0], false, false);
        acc[1][1] = __builtin_amdgcn_wmma_f32_16x16x32_bf16(false, a1, false, b1, (short)0, acc[1][1], false, false);
    }
    #pragma unroll
    for (int i = 0; i < 2; ++i)
        #pragma unroll
        for (int jj = 0; jj < 2; ++jj) {
            const int col = n0 + jj * 16 + l15;
            const float bv = bias ? bias[col] : 0.f;
            #pragma unroll
            for (int r = 0; r < 8; ++r) {
                const int row = m0 + i * 16 + r + kh * 8;
                float v = acc[i][jj][r] + bv;
                if (act == 1) v = (v > 0.f) ? v : 0.01f * v;
                C[(size_t)row * N + col] = v;
                if (Cbf) Cbf[(size_t)row * N + col] = f2bf(v);
            }
        }
}

// ---------------- attention layer 2: sigmoid(h . w2 + b2) per row ----------------
__global__ __launch_bounds__(256)
void k_att2(const float* __restrict__ H, const float* __restrict__ w2,
            const float* __restrict__ b2, float* __restrict__ ew) {
    const int lane = threadIdx.x & 31;
    const int row  = blockIdx.x * 8 + (threadIdx.x >> 5);
    const float4* p = reinterpret_cast<const float4*>(H + (size_t)row * HV) + lane * 2;
    const float4* q = reinterpret_cast<const float4*>(w2) + lane * 2;
    float4 x0 = p[0], x1 = p[1], y0 = q[0], y1 = q[1];
    float s = x0.x*y0.x + x0.y*y0.y + x0.z*y0.z + x0.w*y0.w
            + x1.x*y1.x + x1.y*y1.y + x1.z*y1.z + x1.w*y1.w;
    for (int o = 16; o > 0; o >>= 1) s += __shfl_xor(s, o);
    if (lane == 0) ew[row] = sigm(s + b2[0]);
}

// ---------------- misc elementwise kernels ----------------
__global__ void k_zero(float* __restrict__ p, int n) {
    int i = blockIdx.x * 256 + threadIdx.x;
    if (i < n) p[i] = 0.f;
}
__global__ void k_copy(const float* __restrict__ a, float* __restrict__ b, int n) {
    int i = blockIdx.x * 256 + threadIdx.x;
    if (i < n) b[i] = a[i];
}
__global__ void k_cvt(const float* __restrict__ a, unsigned short* __restrict__ b, int n) {
    int i = blockIdx.x * 256 + threadIdx.x;
    if (i < n) b[i] = f2bf(a[i]);
}
__global__ void k_degree(const int* __restrict__ src2, const int* __restrict__ dst2,
                         const float* __restrict__ ew, float* __restrict__ dr,
                         float* __restrict__ dc) {
    int e = blockIdx.x * 256 + threadIdx.x;
    float w = ew[e];
    atomicAdd(dr + src2[e], w);
    atomicAdd(dc + dst2[e], w);
}
__global__ void k_invsqrt(float* __restrict__ p) {    // over dr|dc contiguous (2*NV)
    int i = blockIdx.x * 256 + threadIdx.x;
    p[i] = 1.f / sqrtf(p[i] + 1e-6f);
}
__global__ void k_scatter(const int* __restrict__ src2, const int* __restrict__ dst2,
                          const float* __restrict__ ew, const float* __restrict__ dr,
                          const float* __restrict__ dc, float* __restrict__ adj) {
    int e = blockIdx.x * 256 + threadIdx.x;
    int s = src2[e], d = dst2[e];
    atomicAdd(adj + (size_t)s * NV + d, dr[s] * ew[e] * dc[d]);
}
__global__ void k_gru(const float* __restrict__ gi, const float* __restrict__ gh,
                      float* __restrict__ hid, unsigned short* __restrict__ hidbf) {
    int t = blockIdx.x * 256 + threadIdx.x;          // NV*HV exact
    int row = t >> 8, c = t & 255;
    size_t b = (size_t)row * G3;
    float r = sigm(gi[b + c]        + gh[b + c]);
    float z = sigm(gi[b + HV + c]   + gh[b + HV + c]);
    float n = tanhf(gi[b + 2*HV + c] + r * gh[b + 2*HV + c]);
    float h = (1.f - z) * n + z * hid[t];
    hid[t] = h;
    hidbf[t] = f2bf(h);
}
__global__ void k_cat(const float* __restrict__ hv, const float* __restrict__ hid,
                      unsigned short* __restrict__ cat) {
    int t = blockIdx.x * 256 + threadIdx.x;          // NV*512 exact
    int row = t >> 9, c = t & 511;
    float v = (c < HV) ? hv[(size_t)row * HV + c] : hid[(size_t)row * HV + (c - HV)];
    cat[t] = f2bf(v);
}

// ---------------- driver ----------------
extern "C" void kernel_launch(void* const* d_in, const int* in_sizes, int n_in,
                              void* d_out, int out_size, void* d_ws, size_t ws_size,
                              hipStream_t stream) {
    (void)in_sizes; (void)n_in; (void)out_size; (void)ws_size;
    const float* hv   = (const float*)d_in[0];
    const float* he   = (const float*)d_in[1];
    const float* vew1 = (const float*)d_in[2];
    const float* vew2 = (const float*)d_in[3];
    const float* aw1  = (const float*)d_in[4];
    const float* ab1  = (const float*)d_in[5];
    const float* aw2  = (const float*)d_in[6];
    const float* ab2  = (const float*)d_in[7];
    const float* gw[4] = {(const float*)d_in[8],  (const float*)d_in[10],
                          (const float*)d_in[12], (const float*)d_in[14]};
    const float* gb[4] = {(const float*)d_in[9],  (const float*)d_in[11],
                          (const float*)d_in[13], (const float*)d_in[15]};
    const float* Wih = (const float*)d_in[16];
    const float* Whh = (const float*)d_in[17];
    const float* bih = (const float*)d_in[18];
    const float* bhh = (const float*)d_in[19];
    const float* rw1 = (const float*)d_in[20];
    const float* rb1 = (const float*)d_in[21];
    const float* rw2 = (const float*)d_in[22];
    const float* rb2 = (const float*)d_in[23];
    float* out = (float*)d_out;

    char* w = (char*)d_ws;
    int*            src2  = (int*)(w + O_SRC);
    int*            dst2  = (int*)(w + O_DST);
    float*          ew    = (float*)(w + O_EW);
    float*          dr    = (float*)(w + O_DR);
    float*          dc    = (float*)(w + O_DC);
    unsigned short* aw1p  = (unsigned short*)(w + O_AW1P);
    unsigned short* gwp   = (unsigned short*)(w + O_GWP);
    unsigned short* wihp  = (unsigned short*)(w + O_WIHP);
    unsigned short* whhp  = (unsigned short*)(w + O_WHHP);
    unsigned short* rw1p  = (unsigned short*)(w + O_RW1P);
    unsigned short* rw2p  = (unsigned short*)(w + O_RW2P);
    float*          adj   = (float*)(w + O_ADJ);
    unsigned short* adjbf = (unsigned short*)(w + O_ADJBF);
    float*          hid   = (float*)(w + O_HID);
    unsigned short* hida  = (unsigned short*)(w + O_HIDA);
    unsigned short* hidp  = (unsigned short*)(w + O_HIDP);
    float*          T     = (float*)(w + O_T);
    unsigned short* Tbf   = (unsigned short*)(w + O_TBF);
    float*          msg   = (float*)(w + O_MSG);
    unsigned short* msgbf = (unsigned short*)(w + O_MSGBF);
    unsigned short* uev   = (unsigned short*)(w + O_UEV);
    float*          hatt  = (float*)(w + O_HATT);
    float*          gi    = (float*)(w + O_GI);
    float*          gh    = (float*)(w + O_GH);
    unsigned short* cat   = (unsigned short*)(w + O_CAT);
    float*          r1    = (float*)(w + O_R1);
    unsigned short* r1bf  = (unsigned short*)(w + O_R1BF);

    const dim3 blk(256);

    // 1) recover edge endpoints from one-hot incidence (coalesced single pass)
    k_extract<<<dim3(((size_t)NV * NE) / 256), blk, 0, stream>>>(vew1, vew2, src2, dst2);

    // 2) pack all weight matrices into WMMA-B fragment layout (bf16)
    k_pack_b<<<dim3((KATT * HV / 2 + 255) / 256), blk, 0, stream>>>(aw1, (unsigned*)aw1p, KATT, HV, 0);
    for (int i = 0; i < 4; ++i)
        k_pack_b<<<dim3((HV * HV / 2 + 255) / 256), blk, 0, stream>>>(
            gw[i], (unsigned*)(gwp + (size_t)i * HV * HV), HV, HV, 0);
    k_pack_b<<<dim3((HV * G3 / 2 + 255) / 256), blk, 0, stream>>>(Wih, (unsigned*)wihp, HV, G3, 1);
    k_pack_b<<<dim3((HV * G3 / 2 + 255) / 256), blk, 0, stream>>>(Whh, (unsigned*)whhp, HV, G3, 1);
    k_pack_b<<<dim3((2 * HV * HV / 2 + 255) / 256), blk, 0, stream>>>(rw1, (unsigned*)rw1p, 2 * HV, HV, 0);
    k_pack_b<<<dim3((HV * OUTD / 2 + 255) / 256), blk, 0, stream>>>(rw2, (unsigned*)rw2p, HV, OUTD, 0);

    // 3) attention MLP: uev -> hatt (leaky) -> ew (sigmoid)
    k_uev<<<dim3(((size_t)E2 * KATT) / 256), blk, 0, stream>>>(hv, he, src2, dst2, uev);
    k_gemm_bf16<<<dim3(HV / 128, E2 / 64), blk, 0, stream>>>(uev, aw1p, ab1, hatt, nullptr, E2, HV, KATT, 1);
    k_att2<<<dim3(E2 / 8), blk, 0, stream>>>(hatt, aw2, ab2, ew);

    // 4) degrees + normalized adjacency (scatter, pre-scaled by dr*dc)
    k_zero<<<dim3(2 * NV / 256), blk, 0, stream>>>(dr, 2 * NV);           // dr|dc contiguous
    k_degree<<<dim3(E2 / 256), blk, 0, stream>>>(src2, dst2, ew, dr, dc);
    k_invsqrt<<<dim3(2 * NV / 256), blk, 0, stream>>>(dr);
    k_zero<<<dim3((size_t)NV * NV / 256), blk, 0, stream>>>(adj, NV * NV);
    k_scatter<<<dim3(E2 / 256), blk, 0, stream>>>(src2, dst2, ew, dr, dc, adj);
    k_cvt<<<dim3((size_t)NV * NV / 256), blk, 0, stream>>>(adj, adjbf, NV * NV);

    // 5) GCN + GRU loop
    k_copy<<<dim3(NV * HV / 256), blk, 0, stream>>>(hv, hid, NV * HV);
    k_cvt<<<dim3(NV * HV / 256), blk, 0, stream>>>(hv, hida, NV * HV);
    for (int it = 0; it < 4; ++it) {
        k_pack_b<<<dim3((NV * HV / 2) / 256), blk, 0, stream>>>(hid, (unsigned*)hidp, NV, HV, 0);
        // T = adj @ hidden      (M=4096, N=256, K=4096)
        k_gemm_bf16<<<dim3(HV / 128, NV / 64), blk, 0, stream>>>(adjbf, hidp, nullptr, T, Tbf, NV, HV, NV, 0);
        // msg = leaky(T @ gw + gb)
        k_gemm_bf16<<<dim3(HV / 128, NV / 64), blk, 0, stream>>>(Tbf, gwp + (size_t)it * HV * HV, gb[it], msg, msgbf, NV, HV, HV, 1);
        // gi = msg @ Wih^T + bih ; gh = hid @ Whh^T + bhh
        k_gemm_bf16<<<dim3(G3 / 128, NV / 64), blk, 0, stream>>>(msgbf, wihp, bih, gi, nullptr, NV, G3, HV, 0);
        k_gemm_bf16<<<dim3(G3 / 128, NV / 64), blk, 0, stream>>>(hida, whhp, bhh, gh, nullptr, NV, G3, HV, 0);
        k_gru<<<dim3(NV * HV / 256), blk, 0, stream>>>(gi, gh, hid, hida);
    }

    // 6) remap MLP: out = (leaky([hv|hid] @ rw1 + rb1)) @ rw2 + rb2
    k_cat<<<dim3(NV * 2 * HV / 256), blk, 0, stream>>>(hv, hid, cat);
    k_gemm_bf16<<<dim3(HV / 128, NV / 64), blk, 0, stream>>>(cat, rw1p, rb1, r1, r1bf, NV, HV, 2 * HV, 1);
    k_gemm_bf16<<<dim3(OUTD / 128, NV / 64), blk, 0, stream>>>(r1bf, rw2p, rb2, out, nullptr, NV, OUTD, HV, 0);
}
